// Involution_43224550867938
// MI455X (gfx1250) — matrix operand order repaired and checked by reading, not compiled, not run on hardware
//
#include <hip/hip_runtime.h>

typedef __attribute__((ext_vector_type(16))) _Float16 v16h;
typedef __attribute__((ext_vector_type(8)))  _Float16 v8h;
typedef __attribute__((ext_vector_type(8)))  float    v8f;

#define KSZ   7
#define PADR  3
#define HH    64
#define WW    64
#define CIN   256
#define CMID  64
#define COUT2 784   // 16 groups * 49 taps
#define NGRP  16
#define GC    16
#define EPSV  1e-5f

#define W1_ELEMS (CMID * CIN)     // 16384
#define W2_ELEMS (COUT2 * CMID)   // 50176

static __device__ __forceinline__ v16h cat8(v8h lo, v8h hi) {
    return __builtin_shufflevector(lo, hi, 0, 1, 2, 3, 4, 5, 6, 7,
                                           8, 9, 10, 11, 12, 13, 14, 15);
}

// ---------------------------------------------------------------------------
// Kernel 0: one-shot f32 -> f16 conversion of both weight matrices into the
// workspace. Removes all per-tile v_cvt work from the GEMM kernels and halves
// the weight fetch bytes (both matrices then live f16 in L2).
// ---------------------------------------------------------------------------
__global__ __launch_bounds__(256) void cvt_weights_kernel(
    const float* __restrict__ w1, const float* __restrict__ w2,
    _Float16* __restrict__ w1h, _Float16* __restrict__ w2h)
{
    const int i = blockIdx.x * 256 + threadIdx.x;
    if (i < W1_ELEMS) w1h[i] = (_Float16)w1[i];
    if (i < W2_ELEMS) w2h[i] = (_Float16)w2[i];
}

// ---------------------------------------------------------------------------
// Kernel 1: t = relu(bn(conv1_w @ x))   -> f16 workspace [b][64][4096]
// One WG = 32 pixels x 64 out channels. 8 waves, each one 16x16 WMMA tile.
// ---------------------------------------------------------------------------
__global__ __launch_bounds__(256) void conv1_bn_relu_kernel(
    const float* __restrict__ x, const _Float16* __restrict__ w1h,
    const float* __restrict__ gamma, const float* __restrict__ beta,
    const float* __restrict__ mean, const float* __restrict__ var,
    _Float16* __restrict__ t_ws)
{
    __shared__ _Float16 xs[32][CIN + 8];   // pixel-major, K contiguous (+pad)

    const int tid  = threadIdx.x;
    const int lane = tid & 31;
    const int wave = tid >> 5;
    const int pixbase = blockIdx.x * 32;
    const int b = blockIdx.y;
    const float* xb = x + (size_t)b * CIN * (HH * WW) + pixbase;

    // Stage x tile (256 K x 32 pixels) to LDS as f16 (coalesced 128B reads).
    {
        const int p = lane;
        #pragma unroll
        for (int i = 0; i < 32; ++i) {
            const int c = wave + 8 * i;
            xs[p][c] = (_Float16)xb[(size_t)c * (HH * WW) + p];
        }
    }
    __syncthreads();

    const int mtile  = wave & 3;     // 4 M-tiles of 16 channels
    const int ntile  = wave >> 2;    // 2 N-tiles of 16 pixels
    const int row    = lane & 15;
    const int hiHalf = (lane >= 16);
    const _Float16* wrow = w1h + (mtile * 16 + row) * CIN;  // A-matrix row
    const _Float16* bro  = &xs[ntile * 16 + row][0];        // B-matrix column

    v8f acc = {};
    #pragma unroll
    for (int kk = 0; kk < CIN; kk += 32) {
        // A 16x32 f16: lanes0-15 K={kk..+7, kk+16..+23}; lanes16-31 +8 each.
        v8h alo = *(const v8h*)(wrow + kk + (hiHalf ? 8 : 0));
        v8h ahi = *(const v8h*)(wrow + kk + (hiHalf ? 24 : 16));
        // B 32x16 f16: same K order as A (lanes0-15 K={kk..+7, kk+16..+23}; lanes16-31 +8).
        const _Float16* bp = bro + kk + (hiHalf ? 8 : 0);
        v8h blo = *(const v8h*)(bp);
        v8h bhi = *(const v8h*)(bp + 16);
        acc = __builtin_amdgcn_wmma_f32_16x16x32_f16(
            false, cat8(alo, ahi), false, cat8(blo, bhi),
            (short)0, acc, false, false);
    }

    // BN + ReLU epilogue; store t as f16 [b][c][hw]
    #pragma unroll
    for (int r = 0; r < 8; ++r) {
        const int oc = mtile * 16 + r + (hiHalf ? 8 : 0);
        const float sc = gamma[oc] * rsqrtf(var[oc] + EPSV);
        const float sh = beta[oc] - mean[oc] * sc;
        float v = acc[r] * sc + sh;
        v = v > 0.f ? v : 0.f;
        t_ws[((size_t)b * CMID + oc) * (HH * WW) + pixbase + ntile * 16 + (lane & 15)]
            = (_Float16)v;
    }
}

// ---------------------------------------------------------------------------
// Kernel 2: fused conv2 (WMMA -> 784 per-pixel weights in LDS) + involution.
// One WG = (batch b, row h, 32-pixel half-row). LDS ~126 KB (CDNA5: 320 KB/WGP).
// ---------------------------------------------------------------------------
#define TS_HALFS   (32 * 72)                 // t tile  [pix][c] f16 (+pad)
#define WT_FLOATS  (COUT2 * 33)              // weights [784][32] f32 (+pad)
#define XS_FLOATS  (GC * KSZ * 40)           // x slab  [16][7][40] f32
#define SMEM_BYTES (TS_HALFS * 2 + WT_FLOATS * 4 + XS_FLOATS * 4)

__global__ __launch_bounds__(256) void conv2_involution_kernel(
    const float* __restrict__ x, const _Float16* __restrict__ t_ws,
    const _Float16* __restrict__ w2h, const float* __restrict__ b2,
    float* __restrict__ out)
{
    extern __shared__ char smem[];
    _Float16* ts  = (_Float16*)smem;                               // [32][72]
    float*    wt  = (float*)(smem + TS_HALFS * 2);                 // [784][33]
    float*    xsl = (float*)(smem + TS_HALFS * 2 + WT_FLOATS * 4); // [16][7][40]

    const int tid  = threadIdx.x;
    const int lane = tid & 31;
    const int wave = tid >> 5;
    const int b     = blockIdx.z;
    const int h     = blockIdx.y;
    const int wbase = blockIdx.x * 32;

    // Stage t tile [64 c][32 pix] -> LDS pixel-major f16
    for (int idx = tid; idx < CMID * 32; idx += 256) {
        const int c = idx >> 5, p = idx & 31;
        ts[p * 72 + c] =
            t_ws[((size_t)b * CMID + c) * (HH * WW) + h * WW + wbase + p];
    }
    __syncthreads();

    // GEMM: wt[784][32] = w2[784][64] @ ttile[64][32] + bias, into LDS.
    // 49 M-tiles x 2 N-tiles = 98 WMMA tiles, wave-strided (EXEC uniform).
    const int row    = lane & 15;
    const int hiHalf = (lane >= 16);
    for (int tileIdx = wave; tileIdx < 98; tileIdx += 8) {
        const int mtile = tileIdx >> 1;
        const int ntile = tileIdx & 1;
        const _Float16* wrow = w2h + (mtile * 16 + row) * CMID; // 784 = 49*16
        const int pix = ntile * 16 + row;
        const _Float16* bro = ts + pix * 72;

        v8f acc = {};
        #pragma unroll
        for (int kk = 0; kk < CMID; kk += 32) {
            v8h alo = *(const v8h*)(wrow + kk + (hiHalf ? 8 : 0));
            v8h ahi = *(const v8h*)(wrow + kk + (hiHalf ? 24 : 16));
            const _Float16* bp = bro + kk + (hiHalf ? 8 : 0);
            v8h blo = *(const v8h*)(bp);
            v8h bhi = *(const v8h*)(bp + 16);
            acc = __builtin_amdgcn_wmma_f32_16x16x32_f16(
                false, cat8(alo, ahi), false, cat8(blo, bhi),
                (short)0, acc, false, false);
        }
        #pragma unroll
        for (int r = 0; r < 8; ++r) {
            const int orow = mtile * 16 + r + (hiHalf ? 8 : 0);
            wt[orow * 33 + pix] = acc[r] + b2[orow];
        }
    }
    __syncthreads();

    // Involution: per group, stage x slab (16ch x 7rows x 38cols, zero-padded)
    // then 49-tap FMA per (channel, pixel).
    const int p2  = tid & 31;   // pixel within tile
    const int cc0 = tid >> 5;   // channel-in-group base (0..7), +8 second half
    for (int g = 0; g < NGRP; ++g) {
        for (int idx = tid; idx < XS_FLOATS; idx += 256) {
            const int col = idx % 40;
            const int kh  = (idx / 40) % KSZ;
            const int cc  = idx / (40 * KSZ);
            const int gw = wbase - PADR + col;
            const int gh = h - PADR + kh;
            float v = 0.f;
            if (col < 38 && gw >= 0 && gw < WW && gh >= 0 && gh < HH)
                v = x[(((size_t)b * CIN + g * GC + cc) * HH + gh) * WW + gw];
            xsl[idx] = v;
        }
        __syncthreads();

        #pragma unroll
        for (int half = 0; half < 2; ++half) {
            const int cc = cc0 + 8 * half;
            float acc = 0.f;
            #pragma unroll
            for (int kh = 0; kh < KSZ; ++kh) {
                #pragma unroll
                for (int kw = 0; kw < KSZ; ++kw) {
                    acc += wt[(g * 49 + kh * 7 + kw) * 33 + p2] *
                           xsl[(cc * KSZ + kh) * 40 + p2 + kw];
                }
            }
            out[(((size_t)b * CIN + g * GC + cc) * HH + h) * WW + wbase + p2] = acc;
        }
        __syncthreads();
    }
}

// ---------------------------------------------------------------------------
extern "C" void kernel_launch(void* const* d_in, const int* in_sizes, int n_in,
                              void* d_out, int out_size, void* d_ws, size_t ws_size,
                              hipStream_t stream) {
    (void)in_sizes; (void)n_in; (void)out_size; (void)ws_size;
    const float* x      = (const float*)d_in[0];
    const float* w1     = (const float*)d_in[1];
    const float* gamma  = (const float*)d_in[2];
    const float* beta   = (const float*)d_in[3];
    const float* mean   = (const float*)d_in[4];
    const float* var    = (const float*)d_in[5];
    const float* w2     = (const float*)d_in[6];
    const float* b2     = (const float*)d_in[7];
    float* out          = (float*)d_out;

    // Workspace layout: t (f16, 2 MB) | w1h (32 KB) | w2h (100 KB)
    _Float16* t_ws = (_Float16*)d_ws;
    _Float16* w1h  = t_ws + (size_t)4 * CMID * HH * WW;
    _Float16* w2h  = w1h + W1_ELEMS;

    (void)hipFuncSetAttribute((const void*)conv2_involution_kernel,
                              hipFuncAttributeMaxDynamicSharedMemorySize,
                              SMEM_BYTES);

    // Kernel 0: weight conversion (once per call; 66K elements)
    cvt_weights_kernel<<<(W2_ELEMS + 255) / 256, 256, 0, stream>>>(
        w1, w2, w1h, w2h);

    // Kernel 1: grid = (4096/32 pixel-tiles, 4 batches)
    conv1_bn_relu_kernel<<<dim3((HH * WW) / 32, 4), 256, 0, stream>>>(
        x, w1h, gamma, beta, mean, var, t_ws);

    // Kernel 2: grid = (2 half-rows, 64 rows, 4 batches)
    conv2_involution_kernel<<<dim3(2, HH, 4), 256, SMEM_BYTES, stream>>>(
        x, t_ws, w2h, b2, out);
}
